// GraphNetwork_GATv2_962072674438
// MI455X (gfx1250) — matrix operand order repaired and checked by reading, compiled, not run on hardware
//
#include <hip/hip_runtime.h>
#include <hip/hip_bf16.h>

// ---------------------------------------------------------------------------
// GATv2 (2 layers) for MI455X / gfx1250.
// GEMMs: v_wmma_f32_16x16x32_f16 (f16 in, f32 accum), 32x64 per wave.
// Edge phase: coalesced wave-per-(edge,head) scoring + float atomics.
// ---------------------------------------------------------------------------

typedef __attribute__((ext_vector_type(16))) _Float16 v16h;
typedef __attribute__((ext_vector_type(8)))  float    v8f;

#define N_NODES 10000
#define N_EDGES 160000
#define N_ETOT  170000   // + self loops
#define F_IN    128
#define D1      512      // heads*hid of layer 1
#define HID     64
#define HEADS   8
#define NEG_SLOPE 0.2f

// ---------------------------------------------------------------------------
// deterministic mean of edge_attr (single block)
// ---------------------------------------------------------------------------
__global__ void mean_kernel(const float* __restrict__ ea, float* __restrict__ out, int n) {
    __shared__ float sm[256];
    float s = 0.f;
    for (int i = threadIdx.x; i < n; i += 256) s += ea[i];
    sm[threadIdx.x] = s;
    __syncthreads();
    for (int off = 128; off > 0; off >>= 1) {
        if (threadIdx.x < off) sm[threadIdx.x] += sm[threadIdx.x + off];
        __syncthreads();
    }
    if (threadIdx.x == 0) *out = sm[0] / (float)n;
}

__global__ void fill_f32(float* __restrict__ p, float v, int n) {
    int t = blockIdx.x * blockDim.x + threadIdx.x;
    if (t < n) p[t] = v;
}
__global__ void fill_u32(unsigned* __restrict__ p, unsigned v, int n) {
    int t = blockIdx.x * blockDim.x + threadIdx.x;
    if (t < n) p[t] = v;
}

// ---------------------------------------------------------------------------
// Pack A (f32 row-major [M,K]) into per-lane WMMA fragments (f16).
// Packed A 16x32 layout (contiguous k, as pack_b):
//   lanes 0-15 : row = lane,    halves 0..15 -> K 0..15
//   lanes16-31 : row = lane-16, halves 0..15 -> K 16..31
// Packed buffer index: ((mt*Kc + kc)*32 + lane) * 16 halves.
// ---------------------------------------------------------------------------
__global__ void pack_a_kernel(const float* __restrict__ X, _Float16* __restrict__ out,
                              int Mtiles, int Kc, int K) {
    int t = blockIdx.x * blockDim.x + threadIdx.x;
    int total = Mtiles * Kc * 32;
    if (t >= total) return;
    int lane = t & 31;
    int kc   = (t >> 5) % Kc;
    int mt   = t / (Kc * 32);
    int row  = mt * 16 + (lane & 15);
    int kbase = kc * 32 + ((lane < 16) ? 0 : 16);
    const float* xr = X + (size_t)row * K;
    v16h v;
#pragma unroll
    for (int e = 0; e < 16; ++e) {
        int j = e >> 1, p = e & 1;
        int ko = 2 * j + p;
        v[e] = (_Float16)xr[kbase + ko];
    }
    ((v16h*)out)[t] = v;
}

// ---------------------------------------------------------------------------
// Pack B = [W0 | W1] (each f32 row-major [K, Nhalf]) into WMMA B fragments.
// Packed B 32x16 layout (contiguous k, as pack_a): lane gives col (lane&15); lanes 0-15 hold
// K 0..15 (sequential halves), lanes 16-31 hold K 16..31.
// Packed buffer index: ((nt*Kc + kc)*32 + lane) * 16 halves.
// ---------------------------------------------------------------------------
__global__ void pack_b_kernel(const float* __restrict__ W0, const float* __restrict__ W1,
                              int Nhalf, _Float16* __restrict__ out, int Ntiles, int Kc) {
    int t = blockIdx.x * blockDim.x + threadIdx.x;
    int total = Ntiles * Kc * 32;
    if (t >= total) return;
    int lane = t & 31;
    int kc   = (t >> 5) % Kc;
    int nt   = t / (Kc * 32);
    int col  = nt * 16 + (lane & 15);
    int kb   = kc * 32 + ((lane < 16) ? 0 : 16);
    const float* W = (col < Nhalf) ? W0 : W1;
    int c = (col < Nhalf) ? col : (col - Nhalf);
    v16h v;
#pragma unroll
    for (int e = 0; e < 16; ++e) {
        v[e] = (_Float16)W[(size_t)(kb + e) * Nhalf + c];
    }
    ((v16h*)out)[t] = v;
}

// ---------------------------------------------------------------------------
// WMMA GEMM: C[M, Ncols] = A[M,K] * B[K,Ncols], fragments pre-packed.
// Block = 128 threads (4 waves). Wave w handles a 32x64 strip: 2 M-tiles
// (mt0, mt0+1) x 4 N-tiles, sharing B fragments -> 8 wmma per k-step on
// 6 b128-load pairs. Odd tail M-tile handled by clamped loads + skipped
// stores (wave-uniform, EXEC stays all-1s through WMMA).
// ---------------------------------------------------------------------------
__global__ void wmma_gemm_kernel(const _Float16* __restrict__ Ap,
                                 const _Float16* __restrict__ Bp,
                                 float* __restrict__ C,
                                 int Mtiles, int Kc, int Ncols) {
    int lane = threadIdx.x & 31;
    int w    = threadIdx.x >> 5;
    int mt0  = (blockIdx.x * 4 + w) * 2;
    if (mt0 >= Mtiles) return;         // uniform per wave
    int  mt1   = mt0 + 1;
    bool has2  = (mt1 < Mtiles);
    int  mt1c  = has2 ? mt1 : mt0;     // clamp loads for odd tail
    int  nt0   = blockIdx.y * 4;

    const v16h* A = (const v16h*)Ap;
    const v16h* B = (const v16h*)Bp;

    v8f c00 = {}, c01 = {}, c02 = {}, c03 = {};
    v8f c10 = {}, c11 = {}, c12 = {}, c13 = {};
    for (int kc = 0; kc < Kc; ++kc) {
        v16h a0 = A[(mt0  * Kc + kc) * 32 + lane];
        v16h a1 = A[(mt1c * Kc + kc) * 32 + lane];
        v16h b0 = B[((nt0 + 0) * Kc + kc) * 32 + lane];
        v16h b1 = B[((nt0 + 1) * Kc + kc) * 32 + lane];
        v16h b2 = B[((nt0 + 2) * Kc + kc) * 32 + lane];
        v16h b3 = B[((nt0 + 3) * Kc + kc) * 32 + lane];
        c00 = __builtin_amdgcn_wmma_f32_16x16x32_f16(false, a0, false, b0, (short)0, c00, false, false);
        c01 = __builtin_amdgcn_wmma_f32_16x16x32_f16(false, a0, false, b1, (short)0, c01, false, false);
        c02 = __builtin_amdgcn_wmma_f32_16x16x32_f16(false, a0, false, b2, (short)0, c02, false, false);
        c03 = __builtin_amdgcn_wmma_f32_16x16x32_f16(false, a0, false, b3, (short)0, c03, false, false);
        c10 = __builtin_amdgcn_wmma_f32_16x16x32_f16(false, a1, false, b0, (short)0, c10, false, false);
        c11 = __builtin_amdgcn_wmma_f32_16x16x32_f16(false, a1, false, b1, (short)0, c11, false, false);
        c12 = __builtin_amdgcn_wmma_f32_16x16x32_f16(false, a1, false, b2, (short)0, c12, false, false);
        c13 = __builtin_amdgcn_wmma_f32_16x16x32_f16(false, a1, false, b3, (short)0, c13, false, false);
    }

    // C/D layout: VGPR r -> row (lane<16 ? r : r+8), col = lane&15
    int rbase0 = mt0 * 16 + ((lane < 16) ? 0 : 8);
    int colb   = nt0 * 16 + (lane & 15);
#pragma unroll
    for (int r = 0; r < 8; ++r) {
        float* crow = C + (size_t)(rbase0 + r) * Ncols + colb;
        crow[0]  = c00[r];
        crow[16] = c01[r];
        crow[32] = c02[r];
        crow[48] = c03[r];
    }
    if (has2) {
        int rbase1 = mt1 * 16 + ((lane < 16) ? 0 : 8);
#pragma unroll
        for (int r = 0; r < 8; ++r) {
            float* crow = C + (size_t)(rbase1 + r) * Ncols + colb;
            crow[0]  = c10[r];
            crow[16] = c11[r];
            crow[32] = c12[r];
            crow[48] = c13[r];
        }
    }
}

// ---------------------------------------------------------------------------
// monotonic float<->uint encoding for atomicMax-based segment max
// ---------------------------------------------------------------------------
__device__ __forceinline__ unsigned enc_f32(float f) {
    unsigned b = __float_as_uint(f);
    return (b & 0x80000000u) ? ~b : (b | 0x80000000u);
}
__device__ __forceinline__ float dec_f32(unsigned e) {
    unsigned b = (e & 0x80000000u) ? (e & 0x7FFFFFFFu) : ~e;
    return __uint_as_float(b);
}

__device__ __forceinline__ void edge_src_dst(int e, const int* __restrict__ ei,
                                             int& src, int& dst) {
    if (e < N_EDGES) { src = ei[e]; dst = ei[N_EDGES + e]; }
    else             { src = dst = e - N_EDGES; }
}

__device__ __forceinline__ float wave_reduce_add(float s) {
#pragma unroll
    for (int m = 16; m > 0; m >>= 1) s += __shfl_xor(s, m, 32);
    return s;
}

// ---------------------------------------------------------------------------
// Layer-1 edge score, wave-per-(edge,head): lanes read 64 contiguous floats
// of xl[src]/xr[dst] (coalesced), shuffle-reduce, lane 0 writes score and
// does the encoded atomicMax (segment max over dst).
// xlr1 layout [N, 1024]: cols 0..511 = xl, 512..1023 = xr.
// ---------------------------------------------------------------------------
__global__ void score1_kernel(const int* __restrict__ ei, const float* __restrict__ eattr,
                              const float* __restrict__ meanp,
                              const float* __restrict__ xlr1,
                              const float* __restrict__ We1, const float* __restrict__ att1,
                              float* __restrict__ score, unsigned* __restrict__ smax) {
    int gw   = (blockIdx.x * blockDim.x + threadIdx.x) >> 5;   // wave id
    int lane = threadIdx.x & 31;
    if (gw >= N_ETOT * HEADS) return;
    int e = gw >> 3, h = gw & 7;
    int src, dst;
    edge_src_dst(e, ei, src, dst);
    float ea = (e < N_EDGES) ? eattr[e] : *meanp;
    const float* xl = xlr1 + (size_t)src * (2 * D1) + h * HID;
    const float* xr = xlr1 + (size_t)dst * (2 * D1) + D1 + h * HID;
    const float* we = We1 + h * HID;
    const float* at = att1 + h * HID;
    float s = 0.f;
#pragma unroll
    for (int c = lane; c < HID; c += 32) {
        float v = xl[c] + xr[c] + ea * we[c];
        v = (v > 0.f) ? v : NEG_SLOPE * v;
        s += at[c] * v;
    }
    s = wave_reduce_add(s);
    if (lane == 0) {
        score[gw] = s;
        atomicMax(&smax[dst * HEADS + h], enc_f32(s));
    }
}

__global__ void alpha1_kernel(const int* __restrict__ ei,
                              float* __restrict__ score,        // in: score, out: alpha
                              const unsigned* __restrict__ smax,
                              float* __restrict__ denom) {
    int t = blockIdx.x * blockDim.x + threadIdx.x;
    if (t >= N_ETOT * HEADS) return;
    int e = t >> 3, h = t & 7;
    int src, dst;
    edge_src_dst(e, ei, src, dst);
    (void)src;
    float m = dec_f32(smax[dst * HEADS + h]);
    float a = expf(score[t] - m);
    score[t] = a;
    atomicAdd(&denom[dst * HEADS + h], a);
}

__global__ void agg1_kernel(const int* __restrict__ ei,
                            const float* __restrict__ alpha,
                            const float* __restrict__ denom,
                            const float* __restrict__ xlr1,
                            float* __restrict__ hout) {
    int t = blockIdx.x * blockDim.x + threadIdx.x;
    if (t >= N_ETOT * D1) return;
    int c = t & 63;
    int h = (t >> 6) & 7;
    int e = t >> 9;
    int src, dst;
    edge_src_dst(e, ei, src, dst);
    float w = alpha[e * HEADS + h] / denom[dst * HEADS + h];
    float msg = w * xlr1[(size_t)src * (2 * D1) + h * HID + c];
    atomicAdd(&hout[(size_t)dst * D1 + h * HID + c], msg);
}

__global__ void bias_kernel(float* __restrict__ p, const float* __restrict__ b,
                            int n, int stride) {
    int t = blockIdx.x * blockDim.x + threadIdx.x;
    if (t < n) p[t] += b[t % stride];
}

// ---------------------------------------------------------------------------
// Layer-2 edge phase (heads=1, ch=64), wave-per-edge.
// xlr2 layout [N,128]: 0..63 xl, 64..127 xr.
// ---------------------------------------------------------------------------
__global__ void score2_kernel(const int* __restrict__ ei, const float* __restrict__ eattr,
                              const float* __restrict__ meanp,
                              const float* __restrict__ xlr2,
                              const float* __restrict__ We2, const float* __restrict__ att2,
                              float* __restrict__ score, unsigned* __restrict__ smax) {
    int e    = (blockIdx.x * blockDim.x + threadIdx.x) >> 5;   // wave id = edge
    int lane = threadIdx.x & 31;
    if (e >= N_ETOT) return;
    int src, dst;
    edge_src_dst(e, ei, src, dst);
    float ea = (e < N_EDGES) ? eattr[e] : *meanp;
    const float* xl = xlr2 + (size_t)src * 128;
    const float* xr = xlr2 + (size_t)dst * 128 + 64;
    float s = 0.f;
#pragma unroll
    for (int c = lane; c < HID; c += 32) {
        float v = xl[c] + xr[c] + ea * We2[c];
        v = (v > 0.f) ? v : NEG_SLOPE * v;
        s += att2[c] * v;
    }
    s = wave_reduce_add(s);
    if (lane == 0) {
        score[e] = s;
        atomicMax(&smax[dst], enc_f32(s));
    }
}

__global__ void alpha2_kernel(const int* __restrict__ ei,
                              float* __restrict__ score,
                              const unsigned* __restrict__ smax,
                              float* __restrict__ denom) {
    int e = blockIdx.x * blockDim.x + threadIdx.x;
    if (e >= N_ETOT) return;
    int src, dst;
    edge_src_dst(e, ei, src, dst);
    (void)src;
    float a = expf(score[e] - dec_f32(smax[dst]));
    score[e] = a;
    atomicAdd(&denom[dst], a);
}

__global__ void agg2_kernel(const int* __restrict__ ei,
                            const float* __restrict__ alpha,
                            const float* __restrict__ denom,
                            const float* __restrict__ xlr2,
                            float* __restrict__ out) {
    int t = blockIdx.x * blockDim.x + threadIdx.x;
    if (t >= N_ETOT * HID) return;
    int c = t & 63;
    int e = t >> 6;
    int src, dst;
    edge_src_dst(e, ei, src, dst);
    float w = alpha[e] / denom[dst];
    atomicAdd(&out[(size_t)dst * HID + c], w * xlr2[(size_t)src * 128 + c]);
}

// ---------------------------------------------------------------------------
// Host launcher
// ---------------------------------------------------------------------------
static inline int ceil_div(long long a, long long b) { return (int)((a + b - 1) / b); }

extern "C" void kernel_launch(void* const* d_in, const int* in_sizes, int n_in,
                              void* d_out, int out_size, void* d_ws, size_t ws_size,
                              hipStream_t stream) {
    const float* x     = (const float*)d_in[0];
    const int*   ei    = (const int*)d_in[1];
    const float* eattr = (const float*)d_in[2];
    const float* Wl1   = (const float*)d_in[3];
    const float* Wr1   = (const float*)d_in[4];
    const float* We1   = (const float*)d_in[5];
    const float* att1  = (const float*)d_in[6];
    const float* b1    = (const float*)d_in[7];
    const float* Wl2   = (const float*)d_in[8];
    const float* Wr2   = (const float*)d_in[9];
    const float* We2   = (const float*)d_in[10];
    const float* att2  = (const float*)d_in[11];
    const float* b2    = (const float*)d_in[12];
    float* out = (float*)d_out;

    // ---- workspace layout (256B aligned sub-allocations) ----
    char* base = (char*)d_ws;
    size_t off = 0;
    auto alloc = [&](size_t bytes) -> char* {
        char* p = base + off;
        off = (off + bytes + 255) & ~(size_t)255;
        return p;
    };
    float*    meanp  = (float*)   alloc(4);
    _Float16* A1     = (_Float16*)alloc((size_t)N_NODES * F_IN * 2);        // 2.56 MB
    _Float16* B1     = (_Float16*)alloc((size_t)F_IN * (2 * D1) * 2);       // 256 KB
    float*    xlr1   = (float*)   alloc((size_t)N_NODES * (2 * D1) * 4);    // 41 MB
    float*    score1 = (float*)   alloc((size_t)N_ETOT * HEADS * 4);        // 5.4 MB
    unsigned* smax1  = (unsigned*)alloc((size_t)N_NODES * HEADS * 4);
    float*    denom1 = (float*)   alloc((size_t)N_NODES * HEADS * 4);
    float*    hbuf   = (float*)   alloc((size_t)N_NODES * D1 * 4);          // 20.5 MB
    _Float16* A2     = (_Float16*)alloc((size_t)N_NODES * D1 * 2);          // 10.2 MB
    _Float16* B2     = (_Float16*)alloc((size_t)D1 * 128 * 2);
    float*    xlr2   = (float*)   alloc((size_t)N_NODES * 128 * 4);         // 5.1 MB
    float*    score2 = (float*)   alloc((size_t)N_ETOT * 4);
    unsigned* smax2  = (unsigned*)alloc((size_t)N_NODES * 4);
    float*    denom2 = (float*)   alloc((size_t)N_NODES * 4);
    (void)ws_size; (void)in_sizes; (void)n_in; (void)out_size;

    const int TB = 256;
    const int Mtiles = N_NODES / 16;  // 625

    // ---- self-loop edge_attr mean ----
    mean_kernel<<<1, 256, 0, stream>>>(eattr, meanp, N_EDGES);

    // ---- layer 1 GEMM: xlr1 = x @ [Wl1 | Wr1]  (M=10000, K=128, N=1024) ----
    {
        int Kc = F_IN / 32;         // 4
        int Ntiles = (2 * D1) / 16; // 64
        pack_a_kernel<<<ceil_div((long long)Mtiles * Kc * 32, TB), TB, 0, stream>>>(
            x, A1, Mtiles, Kc, F_IN);
        pack_b_kernel<<<ceil_div((long long)Ntiles * Kc * 32, TB), TB, 0, stream>>>(
            Wl1, Wr1, D1, B1, Ntiles, Kc);
        dim3 grid(ceil_div(ceil_div(Mtiles, 2), 4), (2 * D1) / 64);  // (79, 16)
        wmma_gemm_kernel<<<grid, 128, 0, stream>>>(A1, B1, xlr1, Mtiles, Kc, 2 * D1);
    }

    // ---- layer 1 edge phase ----
    fill_u32<<<ceil_div(N_NODES * HEADS, TB), TB, 0, stream>>>(smax1, 0u, N_NODES * HEADS);
    fill_f32<<<ceil_div(N_NODES * HEADS, TB), TB, 0, stream>>>(denom1, 0.f, N_NODES * HEADS);
    fill_f32<<<ceil_div(N_NODES * D1, TB), TB, 0, stream>>>(hbuf, 0.f, N_NODES * D1);

    // wave per (edge, head): 32 threads each
    score1_kernel<<<ceil_div((long long)N_ETOT * HEADS * 32, TB), TB, 0, stream>>>(
        ei, eattr, meanp, xlr1, We1, att1, score1, smax1);
    alpha1_kernel<<<ceil_div((long long)N_ETOT * HEADS, TB), TB, 0, stream>>>(
        ei, score1, smax1, denom1);
    agg1_kernel<<<ceil_div((long long)N_ETOT * D1, TB), TB, 0, stream>>>(
        ei, score1, denom1, xlr1, hbuf);
    bias_kernel<<<ceil_div(N_NODES * D1, TB), TB, 0, stream>>>(hbuf, b1, N_NODES * D1, D1);

    // ---- layer 2 GEMM: xlr2 = h @ [Wl2 | Wr2]  (M=10000, K=512, N=128) ----
    {
        int Kc = D1 / 32;          // 16
        int Ntiles = 128 / 16;     // 8
        pack_a_kernel<<<ceil_div((long long)Mtiles * Kc * 32, TB), TB, 0, stream>>>(
            hbuf, A2, Mtiles, Kc, D1);
        pack_b_kernel<<<ceil_div((long long)Ntiles * Kc * 32, TB), TB, 0, stream>>>(
            Wl2, Wr2, HID, B2, Ntiles, Kc);
        dim3 grid(ceil_div(ceil_div(Mtiles, 2), 4), 128 / 64);       // (79, 2)
        wmma_gemm_kernel<<<grid, 128, 0, stream>>>(A2, B2, xlr2, Mtiles, Kc, 128);
    }

    // ---- layer 2 edge phase ----
    fill_u32<<<ceil_div(N_NODES, TB), TB, 0, stream>>>(smax2, 0u, N_NODES);
    fill_f32<<<ceil_div(N_NODES, TB), TB, 0, stream>>>(denom2, 0.f, N_NODES);
    fill_f32<<<ceil_div(N_NODES * HID, TB), TB, 0, stream>>>(out, 0.f, N_NODES * HID);

    // wave per edge
    score2_kernel<<<ceil_div((long long)N_ETOT * 32, TB), TB, 0, stream>>>(
        ei, eattr, meanp, xlr2, We2, att2, score2, smax2);
    alpha2_kernel<<<ceil_div(N_ETOT, TB), TB, 0, stream>>>(ei, score2, smax2, denom2);
    agg2_kernel<<<ceil_div((long long)N_ETOT * HID, TB), TB, 0, stream>>>(
        ei, score2, denom2, xlr2, out);
    bias_kernel<<<ceil_div(N_NODES * HID, TB), TB, 0, stream>>>(out, b2, N_NODES * HID, HID);
}
